// GraphTransformerLayer_85959475462868
// MI455X (gfx1250) — compile-verified, matrix operand order checked by reading, not run on hardware
//
#include <hip/hip_runtime.h>

#define N_NODES 50000
#define N_EDGES 500000
#define DIM     128
#define HEADS   8
#define DPH     16
#define DFF     256
#define EPS_BN  1e-5f

// CDNA5 async global->LDS copy path (ASYNCcnt-tracked) — verified assembling.
#define USE_ASYNC_LDS 1

typedef __attribute__((ext_vector_type(16))) __bf16 v16bf;
typedef __attribute__((ext_vector_type(8)))  float  v8f;

__device__ __forceinline__ void wait_asynccnt0() {
#if defined(__has_builtin) && __has_builtin(__builtin_amdgcn_s_wait_asynccnt)
  __builtin_amdgcn_s_wait_asynccnt(0);
#else
  asm volatile("s_wait_asynccnt 0" ::: "memory");
#endif
}

__device__ __forceinline__ v8f wmma_bf16(const v16bf& a, const v16bf& b,
                                         const v8f& c) {
  return __builtin_amdgcn_wmma_f32_16x16x32_bf16(
      /*neg_a=*/false, a, /*neg_b=*/false, b,
      /*c_mod=*/(short)0, c, /*reuse_a=*/false, /*reuse_b=*/false);
}

// ---------------------------------------------------------------------------
// WMMA GEMM: C[nrows,NCOLS] = A[nrows,KDIM] @ B[KDIM,NCOLS] (+bias)(+res)(ReLU)
// fp32 in memory, bf16 WMMA with fp32 accumulate.
// Block = 256 threads = 8 waves; block tile = 32 rows x 128 cols.
// Wave w: rows (w>>2)*16, cols (w&3)*32 -> two 16x16 tiles sharing one A frag.
// K fully unrolled in chunks of 32; B tile (32xKchunk x 128) staged in LDS via
// global_load_async_to_lds_b128, stride 130 (conflict-free: 130r%64 = 2r).
//
// ISA register layouts (CDNA5 ISA):
//  A 16x32 bf16 : lane L holds row m=L%16; khalf=(L/16)*8;
//                 elems 0..7 -> K=khalf+0..7 ; elems 8..15 -> K=16+khalf+0..7
//  B 32x16 bf16 : lane L holds col n=L%16; same K pattern
//  C 16x16 f32  : VGPR r, lane L -> (M = r + 8*(L/16), N = L%16)
// ---------------------------------------------------------------------------
template <int KDIM, int NCOLS, bool HAS_BIAS, bool HAS_RES, bool RELU>
__global__ __launch_bounds__(256) void gemm_wmma_bf16(
    const float* __restrict__ A, const float* __restrict__ B,
    const float* __restrict__ bias, const float* __restrict__ res,
    float* __restrict__ C, int nrows) {
  constexpr int BSTRIDE = 130;
  __shared__ float Bs[32 * BSTRIDE];

  const int tid  = threadIdx.x;
  const int lane = tid & 31;
  const int wave = tid >> 5;

  const int rowBase    = blockIdx.x * 32 + (wave >> 2) * 16;
  const int colBase128 = blockIdx.y * 128;
  const int colLocal   = (wave & 3) * 32;  // within 128-wide block tile

  const int m     = lane & 15;
  const int khalf = (lane >> 4) << 3;      // 0 or 8

  int arow = rowBase + m;
  if (arow >= nrows) arow = nrows - 1;     // clamp; store is guarded
  const float* __restrict__ Aptr = A + arow * KDIM;

  // Cooperative B staging: thread t moves 16 floats of row (t>>3), col (t&7)*16.
  const int brow = tid >> 3;               // 0..31
  const int bc   = (tid & 7) * 16;         // 0..112
  const float* __restrict__ Bsrc0 = B + brow * NCOLS + colBase128 + bc;
  float* const Bdst = &Bs[brow * BSTRIDE + bc];
#if USE_ASYNC_LDS
  const unsigned int ldsoff = (unsigned int)(unsigned long long)Bdst;
#endif
  // Per-lane LDS read bases for the wave's two column tiles.
  const float* const bs0 = &Bs[colLocal + m];
  const float* const bs1 = bs0 + 16;

  v8f acc0 = {}, acc1 = {};
#pragma unroll
  for (int kk = 0; kk < KDIM; kk += 32) {
    __syncthreads();  // protect Bs against previous iteration's readers
    const float* src = Bsrc0 + kk * NCOLS;
#if USE_ASYNC_LDS
    asm volatile(
        "global_load_async_to_lds_b128 %0, %1, off\n\t"
        "global_load_async_to_lds_b128 %0, %1, off offset:16\n\t"
        "global_load_async_to_lds_b128 %0, %1, off offset:32\n\t"
        "global_load_async_to_lds_b128 %0, %1, off offset:48"
        :
        : "v"(ldsoff), "v"((unsigned long long)(uintptr_t)src)
        : "memory");
    wait_asynccnt0();
#else
#pragma unroll
    for (int i = 0; i < 16; ++i) Bdst[i] = src[i];
#endif
    __syncthreads();

    v16bf a, b0, b1;
#pragma unroll
    for (int e = 0; e < 8; ++e) {
      a[e]      = (__bf16)Aptr[kk + khalf + e];
      a[e + 8]  = (__bf16)Aptr[kk + 16 + khalf + e];
      b0[e]     = (__bf16)bs0[(khalf + e) * BSTRIDE];
      b0[e + 8] = (__bf16)bs0[(16 + khalf + e) * BSTRIDE];
      b1[e]     = (__bf16)bs1[(khalf + e) * BSTRIDE];
      b1[e + 8] = (__bf16)bs1[(16 + khalf + e) * BSTRIDE];
    }
    acc0 = wmma_bf16(a, b0, acc0);
    acc1 = wmma_bf16(a, b1, acc1);
  }

  const int rofs  = (lane >> 4) << 3;
  const int ncol0 = colBase128 + colLocal + m;
  const int idx0  = (rowBase + rofs) * NCOLS + ncol0;  // fits in int32
  float bv0 = 0.0f, bv1 = 0.0f;
  if (HAS_BIAS) { bv0 = bias[ncol0]; bv1 = bias[ncol0 + 16]; }

  if (rowBase + 16 <= nrows) {
    // fast path: whole 16-row tile in range, unguarded strided stores
#pragma unroll
    for (int r = 0; r < 8; ++r) {
      float v0 = acc0[r], v1 = acc1[r];
      if (HAS_BIAS) { v0 += bv0; v1 += bv1; }
      if (HAS_RES)  { v0 += res[idx0 + r * NCOLS]; v1 += res[idx0 + r * NCOLS + 16]; }
      if (RELU)     { v0 = fmaxf(v0, 0.0f); v1 = fmaxf(v1, 0.0f); }
      C[idx0 + r * NCOLS]      = v0;
      C[idx0 + r * NCOLS + 16] = v1;
    }
  } else {
#pragma unroll
    for (int r = 0; r < 8; ++r) {
      if (rowBase + rofs + r < nrows) {
        float v0 = acc0[r], v1 = acc1[r];
        if (HAS_BIAS) { v0 += bv0; v1 += bv1; }
        if (HAS_RES)  { v0 += res[idx0 + r * NCOLS]; v1 += res[idx0 + r * NCOLS + 16]; }
        if (RELU)     { v0 = fmaxf(v0, 0.0f); v1 = fmaxf(v1, 0.0f); }
        C[idx0 + r * NCOLS]      = v0;
        C[idx0 + r * NCOLS + 16] = v1;
      }
    }
  }
}

// ---------------------------------------------------------------------------
// Edge phase 1: per (edge, head) attention score -> exp -> segment sum.
// Scores are clipped to [-5,5] so exp() is stable without the segment-max
// pass (alpha = exp(s)/sum exp(s) is mathematically identical).
// ---------------------------------------------------------------------------
__global__ __launch_bounds__(256) void edge_score(
    const float* __restrict__ Q, const float* __restrict__ Km,
    const long long* __restrict__ ei, float* __restrict__ ex,
    float* __restrict__ segsum) {
  const int t = blockIdx.x * blockDim.x + threadIdx.x;
  if (t >= N_EDGES * HEADS) return;
  const int e = t >> 3;
  const int h = t & 7;
  const int s = (int)ei[e];             // src (row 0 of edge_index)
  const int d = (int)ei[N_EDGES + e];   // dst (row 1)

  const float4* q = (const float4*)(Q  + d * DIM + h * DPH);
  const float4* k = (const float4*)(Km + s * DIM + h * DPH);
  float acc = 0.0f;
#pragma unroll
  for (int i = 0; i < 4; ++i) {
    const float4 qa = q[i];
    const float4 ka = k[i];
    acc += qa.x * ka.x + qa.y * ka.y + qa.z * ka.z + qa.w * ka.w;
  }
  acc *= 0.25f;                                     // 1/sqrt(DPH=16)
  acc = fminf(5.0f, fmaxf(-5.0f, acc));
  const float v = __expf(acc);
  ex[t] = v;
  atomicAdd(&segsum[d * HEADS + h], v);
}

// ---------------------------------------------------------------------------
// Edge phase 2: alpha-weighted message scatter-add.
// ---------------------------------------------------------------------------
__global__ __launch_bounds__(256) void edge_agg(
    const float* __restrict__ V, const long long* __restrict__ ei,
    const float* __restrict__ ex, const float* __restrict__ segsum,
    float* __restrict__ agg) {
  const int t = blockIdx.x * blockDim.x + threadIdx.x;
  if (t >= N_EDGES * HEADS) return;
  const int e = t >> 3;
  const int h = t & 7;
  const int s = (int)ei[e];
  const int d = (int)ei[N_EDGES + e];

  const float alpha = ex[t] / (segsum[d * HEADS + h] + 1e-16f);
  const float4* vp = (const float4*)(V + s * DIM + h * DPH);
  float* op = agg + d * DIM + h * DPH;
#pragma unroll
  for (int i = 0; i < 4; ++i) {
    const float4 va = vp[i];
    atomicAdd(&op[i * 4 + 0], va.x * alpha);
    atomicAdd(&op[i * 4 + 1], va.y * alpha);
    atomicAdd(&op[i * 4 + 2], va.z * alpha);
    atomicAdd(&op[i * 4 + 3], va.w * alpha);
  }
}

// ---------------------------------------------------------------------------
// BatchNorm: stats pass (sum, sumsq per feature; lane == feature column so
// loads coalesce), then normalize pass.
// ---------------------------------------------------------------------------
#define BN_ROWS 256
__global__ __launch_bounds__(DIM) void bn_stats(
    const float* __restrict__ h, float* __restrict__ st, int nrows) {
  const int d  = threadIdx.x;          // feature, blockDim.x == DIM
  const int r0 = blockIdx.x * BN_ROWS;
  const int r1 = (r0 + BN_ROWS < nrows) ? (r0 + BN_ROWS) : nrows;
  float s = 0.0f, s2 = 0.0f;
  for (int r = r0; r < r1; ++r) {
    const float v = h[r * DIM + d];
    s += v;
    s2 += v * v;
  }
  atomicAdd(&st[d], s);
  atomicAdd(&st[DIM + d], s2);
}

__global__ __launch_bounds__(256) void bn_apply(
    const float* __restrict__ h, const float* __restrict__ st,
    const float* __restrict__ g, const float* __restrict__ b,
    float* __restrict__ out, int nrows) {
  const int idx = blockIdx.x * blockDim.x + threadIdx.x;
  if (idx >= nrows * DIM) return;
  const int d = idx & (DIM - 1);
  const float inv_n = 1.0f / (float)nrows;
  const float mu  = st[d] * inv_n;
  const float var = st[DIM + d] * inv_n - mu * mu;
  out[idx] = g[d] * (h[idx] - mu) * __frsqrt_rn(var + EPS_BN) + b[d];
}

// ---------------------------------------------------------------------------
// Launch
// ---------------------------------------------------------------------------
extern "C" void kernel_launch(void* const* d_in, const int* in_sizes, int n_in,
                              void* d_out, int out_size, void* d_ws,
                              size_t ws_size, hipStream_t stream) {
  const float*     x   = (const float*)d_in[0];
  const long long* ei  = (const long long*)d_in[1];
  const float*     WQ  = (const float*)d_in[2];
  const float*     WK  = (const float*)d_in[3];
  const float*     WV  = (const float*)d_in[4];
  const float*     WO  = (const float*)d_in[5];
  const float*     bO  = (const float*)d_in[6];
  const float*     W1  = (const float*)d_in[7];
  const float*     b1  = (const float*)d_in[8];
  const float*     W2  = (const float*)d_in[9];
  const float*     b2  = (const float*)d_in[10];
  const float*     g1  = (const float*)d_in[11];
  const float*     be1 = (const float*)d_in[12];
  const float*     g2  = (const float*)d_in[13];
  const float*     be2 = (const float*)d_in[14];
  float*           out = (float*)d_out;

  const size_t ND = (size_t)N_NODES * DIM;
  float* ws   = (float*)d_ws;
  float* Qb   = ws;                                  // ND    (later: h post-WO)
  float* Kb   = Qb + ND;                             // ND    (later: h post-BN1)
  float* Vb   = Kb + ND;                             // ND    (later: h post-FFN2)
  float* agg  = Vb + ND;                             // ND
  float* ffb  = agg + ND;                            // N*DFF = 2*ND
  float* exb  = ffb + 2 * ND;                        // E*H
  float* seg  = exb + (size_t)N_EDGES * HEADS;       // N*H
  float* st   = seg + (size_t)N_NODES * HEADS;       // 2*DIM

  const dim3 blk(256);
  const int  rblocks = (N_NODES + 31) / 32;
  const dim3 gD(rblocks, 1);                         // 128-col GEMMs
  const dim3 gF(rblocks, 2);                         // 256-col GEMM
  const int  ethreads = N_EDGES * HEADS;
  const int  eblocks  = (ethreads + 255) / 256;
  const int  bnblocks = (N_NODES + BN_ROWS - 1) / BN_ROWS;
  const int  apblocks = ((int)ND + 255) / 256;

  // Q, K, V projections (no bias)
  gemm_wmma_bf16<DIM, DIM, false, false, false><<<gD, blk, 0, stream>>>(
      x, WQ, nullptr, nullptr, Qb, N_NODES);
  gemm_wmma_bf16<DIM, DIM, false, false, false><<<gD, blk, 0, stream>>>(
      x, WK, nullptr, nullptr, Kb, N_NODES);
  gemm_wmma_bf16<DIM, DIM, false, false, false><<<gD, blk, 0, stream>>>(
      x, WV, nullptr, nullptr, Vb, N_NODES);

  // Edge attention: scores + segment softmax denominators
  hipMemsetAsync(seg, 0, (size_t)N_NODES * HEADS * sizeof(float), stream);
  edge_score<<<eblocks, 256, 0, stream>>>(Qb, Kb, ei, exb, seg);

  // Weighted scatter-add aggregation
  hipMemsetAsync(agg, 0, ND * sizeof(float), stream);
  edge_agg<<<eblocks, 256, 0, stream>>>(Vb, ei, exb, seg, agg);

  // Output projection + bias + residual(x)  -> Qb (Q is dead)
  gemm_wmma_bf16<DIM, DIM, true, true, false><<<gD, blk, 0, stream>>>(
      agg, WO, bO, x, Qb, N_NODES);

  // BatchNorm 1 -> Kb (K is dead)
  hipMemsetAsync(st, 0, 2 * DIM * sizeof(float), stream);
  bn_stats<<<bnblocks, DIM, 0, stream>>>(Qb, st, N_NODES);
  bn_apply<<<apblocks, 256, 0, stream>>>(Qb, st, g1, be1, Kb, N_NODES);

  // FFN: relu(h@W1+b1) -> ffb ; @W2+b2 + residual(Kb) -> Vb (V is dead)
  gemm_wmma_bf16<DIM, DFF, true, false, true><<<gF, blk, 0, stream>>>(
      Kb, W1, b1, nullptr, ffb, N_NODES);
  gemm_wmma_bf16<DFF, DIM, true, true, false><<<gD, blk, 0, stream>>>(
      ffb, W2, b2, Kb, Vb, N_NODES);

  // BatchNorm 2 -> out
  hipMemsetAsync(st, 0, 2 * DIM * sizeof(float), stream);
  bn_stats<<<bnblocks, DIM, 0, stream>>>(Vb, st, N_NODES);
  bn_apply<<<apblocks, 256, 0, stream>>>(Vb, st, g2, be2, out, N_NODES);
}
